// FastAttention_27075473834157
// MI455X (gfx1250) — matrix operand order repaired and checked by reading, not compiled, not run on hardware
//
#include <hip/hip_runtime.h>
#include <hip/hip_bf16.h>

// Problem constants (from reference setup_inputs)
#define B_ 4
#define L_ 4096
#define H_ 16
#define D_ 64      // head dim == E
#define M_ 256     // feature dim
#define NCHUNK 8
#define SCHUNK (L_ / NCHUNK)   // 512 sequence rows per kv_kernel workgroup

#define SCALE_ 0.35355339059327373f   // 64^-0.25
#define KEPS_ 1e-3f
#define ZEPS_ 1e-6f

typedef __attribute__((ext_vector_type(16))) __bf16 v16bf;
typedef __attribute__((ext_vector_type(8)))  float  v8f;

union FragU { v16bf v; unsigned short u[16]; };

__device__ __forceinline__ unsigned short f2bf(float f) {
  unsigned int x = __float_as_uint(f);
  x += 0x7FFFu + ((x >> 16) & 1u);          // round-to-nearest-even
  return (unsigned short)(x >> 16);
}
__device__ __forceinline__ float bf2f(unsigned short u) {
  return __uint_as_float(((unsigned int)u) << 16);
}
__device__ __forceinline__ v8f vzero8() {
  v8f r;
#pragma unroll
  for (int i = 0; i < 8; ++i) r[i] = 0.0f;
  return r;
}

// A-matrix fragment (16x32 bf16, rows=M dim, K contiguous in src rows).
// Lane l (r=l&15, g=l>>4): u[0..7] = K = g*8 + 0..7 ; u[8..15] = K = 16+g*8+0..7
__device__ __forceinline__ v16bf load_a(const unsigned short* s, int pitch,
                                        int row0, int k0, int lane) {
  const int r = lane & 15, g = lane >> 4;
  const unsigned short* p = s + (row0 + r) * pitch + k0 + g * 8;
  FragU f;
#pragma unroll
  for (int i = 0; i < 8; ++i) f.u[i] = p[i];
#pragma unroll
  for (int i = 0; i < 8; ++i) f.u[8 + i] = p[16 + i];
  return f.v;
}

// B-matrix fragment (32x16 bf16). Source staged K-contiguous as [n][k].
// Lane l (n=l&15, g=l>>4): u[0..7] = K = g*8 + 0..7 ; u[8..15] = K = 16+g*8+0..7
__device__ __forceinline__ v16bf load_b(const unsigned short* s, int pitch,
                                        int n0, int k0, int lane) {
  const int n = lane & 15, g = lane >> 4;
  const unsigned short* p = s + (n0 + n) * pitch + k0 + g * 8;
  FragU f;
#pragma unroll
  for (int i = 0; i < 8; ++i) f.u[i] = p[i];
#pragma unroll
  for (int i = 0; i < 8; ++i) f.u[8 + i] = p[16 + i];
  return f.v;
}

#define WMMA_BF16(a, b, c) \
  __builtin_amdgcn_wmma_f32_16x16x32_bf16(false, (a), false, (b), (short)0, (c), false, false)

// ---------------------------------------------------------------------------
__global__ __launch_bounds__(256) void zero_kernel(float* p, int n) {
  int i = blockIdx.x * blockDim.x + threadIdx.x;
  if (i < n) p[i] = 0.0f;
}

// ---------------------------------------------------------------------------
// Kernel 1: Kf = relu(scale * k @ P^T) + eps ; KV += Kf^T @ v ; Ksum += sum_s Kf
// Grid: B*H*NCHUNK workgroups, 256 threads (8 waves).
__global__ __launch_bounds__(256) void kv_kernel(const float* __restrict__ kg,
                                                 const float* __restrict__ vg,
                                                 const float* __restrict__ proj,
                                                 float* __restrict__ kv,    // [BH, M, D]
                                                 float* __restrict__ ksum)  // [BH, M]
{
  __shared__ unsigned short Pl[M_ * 72];      // P[m][d], pitch 72 (bank-friendly)
  __shared__ unsigned short ktile[32 * 72];   // k[s][d]
  __shared__ unsigned short vT[D_ * 40];      // v^T[e][s]
  __shared__ unsigned short kfT[M_ * 32];     // Kf^T[m][s]

  const int tid  = threadIdx.x;
  const int lane = tid & 31;
  const int w    = tid >> 5;
  const int chunk = blockIdx.x % NCHUNK;
  const int bh    = blockIdx.x / NCHUNK;
  const int b = bh / H_, h = bh % H_;

  // Stage P (native [M][D] layout is exactly the K-contiguous B layout we need)
  {
    const float* pr = proj + tid * D_;
#pragma unroll 8
    for (int d = 0; d < D_; ++d) Pl[tid * 72 + d] = f2bf(pr[d]);
  }

  v8f acc[8];                    // KV tiles: 2 m-tiles x 4 e-tiles per wave
#pragma unroll
  for (int i = 0; i < 8; ++i) acc[i] = vzero8();
  float ksum_reg = 0.0f;

  const int s_begin = chunk * SCHUNK;
  for (int s0 = s_begin; s0 < s_begin + SCHUNK; s0 += 32) {
    __syncthreads();  // previous iteration's consumers of ktile/vT/kfT done

    // Stage 32 rows of k and v (v transposed so its B-fragments are K-contiguous)
    {
      const int row = tid >> 3;           // 0..31
      const int c0  = (tid & 7) * 8;
      const size_t base = ((size_t)(b * L_ + s0 + row) * H_ + h) * D_ + c0;
      const float* kr = kg + base;
      const float* vr = vg + base;
#pragma unroll
      for (int i = 0; i < 8; ++i) ktile[row * 72 + c0 + i] = f2bf(kr[i]);
#pragma unroll
      for (int i = 0; i < 8; ++i) vT[(c0 + i) * 40 + row] = f2bf(vr[i]);
    }
    __syncthreads();

    // Feature map: wave w owns m in [w*32, w*32+32). 4 tiles x K=64 (2 steps).
#pragma unroll
    for (int st = 0; st < 2; ++st) {
#pragma unroll
      for (int mt = 0; mt < 2; ++mt) {
        v8f c = vzero8();
#pragma unroll
        for (int k0 = 0; k0 < 64; k0 += 32) {
          v16bf a  = load_a(ktile, 72, st * 16, k0, lane);
          v16bf bb = load_b(Pl, 72, w * 32 + mt * 16, k0, lane);
          c = WMMA_BF16(a, bb, c);
        }
        // relu(scale*x)+eps, store transposed [m][s] for reuse as A-matrix
        const int m  = w * 32 + mt * 16 + (lane & 15);
        const int sb = st * 16 + ((lane >> 4) << 3);
#pragma unroll
        for (int r = 0; r < 8; ++r) {
          float val = c[r] * SCALE_;
          val = (val > 0.0f ? val : 0.0f) + KEPS_;
          kfT[m * 32 + sb + r] = f2bf(val);
        }
      }
    }
    __syncthreads();

    // Ksum partial: thread tid owns feature m = tid
    {
      const unsigned short* rp = &kfT[tid * 32];
#pragma unroll
      for (int ss = 0; ss < 32; ++ss) ksum_reg += bf2f(rp[ss]);
    }

    // KV accumulation: KV[m,e] += sum_s Kf[s,m] * v[s,e]   (A = Kf^T, B = v)
    {
      v16bf a0 = load_a(kfT, 32, w * 32, 0, lane);
      v16bf a1 = load_a(kfT, 32, w * 32 + 16, 0, lane);
#pragma unroll
      for (int et = 0; et < 4; ++et) {
        v16bf bb = load_b(vT, 40, et * 16, 0, lane);
        acc[et]     = WMMA_BF16(a0, bb, acc[et]);
        acc[4 + et] = WMMA_BF16(a1, bb, acc[4 + et]);
      }
    }
  }

  // Flush partial KV / Ksum with fp32 global atomics (8-way contention per bh)
  float* kvbh = kv + (size_t)bh * (M_ * D_);
#pragma unroll
  for (int mt = 0; mt < 2; ++mt) {
    const int mb = w * 32 + mt * 16 + ((lane >> 4) << 3);
    const int e  = lane & 15;
#pragma unroll
    for (int et = 0; et < 4; ++et) {
#pragma unroll
      for (int r = 0; r < 8; ++r) {
        (void)__hip_atomic_fetch_add(&kvbh[(size_t)(mb + r) * D_ + et * 16 + e],
                                     acc[mt * 4 + et][r],
                                     __ATOMIC_RELAXED, __HIP_MEMORY_SCOPE_AGENT);
      }
    }
  }
  (void)__hip_atomic_fetch_add(&ksum[(size_t)bh * M_ + tid], ksum_reg,
                               __ATOMIC_RELAXED, __HIP_MEMORY_SCOPE_AGENT);
}

// ---------------------------------------------------------------------------
// Kernel 2: Qf = relu(scale * q @ P^T) + eps ; out = (Qf @ KV) / (Qf . Ksum)
// Grid: B*H*(L/64) workgroups, 256 threads. Each WG: 64 L-rows.
__global__ __launch_bounds__(256) void out_kernel(const float* __restrict__ qg,
                                                  const float* __restrict__ proj,
                                                  const float* __restrict__ kv,
                                                  const float* __restrict__ ksum,
                                                  float* __restrict__ outg)
{
  __shared__ unsigned short Pl[M_ * 72];      // P[m][d]
  __shared__ unsigned short qtile[64 * 72];   // q[l][d]
  __shared__ unsigned short QfL[64 * 40];     // Qf[l][m-chunk of 32]
  __shared__ unsigned short kvT[D_ * 40];     // KV^T[e][m-chunk of 32]
  __shared__ float ksumL[M_];
  __shared__ float zL[64];

  const int tid = threadIdx.x, lane = tid & 31, w = tid >> 5;
  const int ltile = blockIdx.x % (L_ / 64);
  const int bh    = blockIdx.x / (L_ / 64);
  const int b = bh / H_, h = bh % H_;
  const int l0 = ltile * 64;

  {
    const float* pr = proj + tid * D_;
#pragma unroll 8
    for (int d = 0; d < D_; ++d) Pl[tid * 72 + d] = f2bf(pr[d]);
  }
  ksumL[tid] = ksum[(size_t)bh * M_ + tid] + ZEPS_;
  {
    const int row = tid >> 2, c0 = (tid & 3) * 16;
    const float* qr = qg + ((size_t)(b * L_ + l0 + row) * H_ + h) * D_ + c0;
#pragma unroll
    for (int i = 0; i < 16; ++i) qtile[row * 72 + c0 + i] = f2bf(qr[i]);
  }

  float zreg = 0.0f;
  v8f oacc[2] = { vzero8(), vzero8() };     // wave: l-tile (w&3) x 32 e-cols ((w>>2)*32)
  const int lt = w & 3, eh = w >> 2;

  for (int mc = 0; mc < 8; ++mc) {
    __syncthreads();  // previous chunk's consumers of QfL/kvT done

    // Stage KV chunk transposed: kvT[e][mloc]
    {
      const int mloc = tid >> 3, e0 = (tid & 7) * 8;
      const float* kr = kv + ((size_t)bh * M_ + mc * 32 + mloc) * D_ + e0;
#pragma unroll
      for (int i = 0; i < 8; ++i) kvT[(e0 + i) * 40 + mloc] = f2bf(kr[i]);
    }

    // Qf chunk: wave w computes l-tile (w&3) x m-tile (w>>2) of this 64x32 slab
    {
      const int mtn = w >> 2;               // 0..1
      v8f c = vzero8();
#pragma unroll
      for (int k0 = 0; k0 < 64; k0 += 32) {
        v16bf a  = load_a(qtile, 72, (w & 3) * 16, k0, lane);
        v16bf bb = load_b(Pl, 72, mc * 32 + mtn * 16, k0, lane);
        c = WMMA_BF16(a, bb, c);
      }
      const int mcol = mtn * 16 + (lane & 15);
      const int lb   = (w & 3) * 16 + ((lane >> 4) << 3);
#pragma unroll
      for (int r = 0; r < 8; ++r) {
        float val = c[r] * SCALE_;
        val = (val > 0.0f ? val : 0.0f) + KEPS_;
        QfL[(lb + r) * 40 + mcol] = f2bf(val);
      }
    }
    __syncthreads();

    // Z partial: thread t<64 owns row l=t (fp32, uses the same bf16-rounded Qf)
    if (tid < 64) {
      const unsigned short* rp = &QfL[tid * 40];
#pragma unroll
      for (int mm = 0; mm < 32; ++mm) zreg += bf2f(rp[mm]) * ksumL[mc * 32 + mm];
    }

    // out accumulation: A = Qf chunk (K=32), B = KV chunk
    {
      v16bf a = load_a(QfL, 40, lt * 16, 0, lane);
#pragma unroll
      for (int et = 0; et < 2; ++et) {
        v16bf bb = load_b(kvT, 40, eh * 32 + et * 16, 0, lane);
        oacc[et] = WMMA_BF16(a, bb, oacc[et]);
      }
    }
  }

  if (tid < 64) zL[tid] = 1.0f / zreg;
  __syncthreads();

  // Scaled store: out[b, l0+ll, h, e]
  {
    const int lb = lt * 16 + ((lane >> 4) << 3);
    const int e  = eh * 32 + (lane & 15);
#pragma unroll
    for (int et = 0; et < 2; ++et) {
#pragma unroll
      for (int r = 0; r < 8; ++r) {
        const int ll = lb + r;
        outg[((size_t)(b * L_ + l0 + ll) * H_ + h) * D_ + e + et * 16] =
            oacc[et][r] * zL[ll];
      }
    }
  }
}

// ---------------------------------------------------------------------------
extern "C" void kernel_launch(void* const* d_in, const int* in_sizes, int n_in,
                              void* d_out, int out_size, void* d_ws, size_t ws_size,
                              hipStream_t stream) {
  const float* q    = (const float*)d_in[0];
  const float* k    = (const float*)d_in[1];
  const float* v    = (const float*)d_in[2];
  const float* proj = (const float*)d_in[3];
  float* out = (float*)d_out;

  float* kv   = (float*)d_ws;                         // [B*H, M, D] fp32 (4 MB)
  float* ksum = kv + (size_t)B_ * H_ * M_ * D_;       // [B*H, M]    (64 KB)

  const int nz = B_ * H_ * M_ * D_ + B_ * H_ * M_;
  zero_kernel<<<(nz + 255) / 256, 256, 0, stream>>>(kv, nz);
  kv_kernel<<<B_ * H_ * NCHUNK, 256, 0, stream>>>(k, v, proj, kv, ksum);
  out_kernel<<<B_ * H_ * (L_ / 64), 256, 0, stream>>>(q, proj, kv, ksum, out);
}
